// QuantLinearWithWeight_19765439496401
// MI455X (gfx1250) — 8-bit matrix operand order repaired and checked by reading, compiled, not run on hardware
//
#include <hip/hip_runtime.h>
#include <stdint.h>

#define M_TOT 65536
#define K_DIM 1024
#define N_DIM 1024

typedef __attribute__((ext_vector_type(8))) int v8i;
typedef __attribute__((ext_vector_type(4))) float vf4;

#define BM 128
#define BN 128
#define BK 64
#define NTHREADS 256
#define KSTEPS (K_DIM / BK)

#if defined(__has_builtin)
#if __has_builtin(__builtin_amdgcn_global_load_async_to_lds_b128)
#define HAVE_ASYNC_LDS 1
#endif
#endif

#ifdef HAVE_ASYNC_LDS
typedef int vi4 __attribute__((vector_size(4 * sizeof(int))));
typedef __attribute__((address_space(1))) vi4* gvi4p;
typedef __attribute__((address_space(3))) vi4* lvi4p;
#endif

__device__ __forceinline__ void copy16_g2l(const signed char* g, signed char* l) {
#ifdef HAVE_ASYNC_LDS
  __builtin_amdgcn_global_load_async_to_lds_b128((gvi4p)(void*)g, (lvi4p)(void*)l,
                                                  0,  0);
#else
  *(int4*)l = *(const int4*)g;
#endif
}

__device__ __forceinline__ void wait_stage() {
#ifdef HAVE_ASYNC_LDS
  asm volatile("s_wait_asynccnt 0" ::: "memory");
#endif
  __syncthreads();
}

__global__ void prep_scales(const float* __restrict__ s_w,
                            const float* __restrict__ b_int8,
                            const float* __restrict__ s_x_ptr,
                            float* __restrict__ sa_out,
                            float* __restrict__ bq_out) {
  int o = blockIdx.x * blockDim.x + threadIdx.x;
  if (o >= N_DIM) return;
  const float sx = s_x_ptr[0];
  sa_out[o] = s_w[o] * sx;
  bq_out[o] = rintf(b_int8[o] / sx);
}

__global__ void quant_pack_f32_to_i8(const float* __restrict__ src,
                                     const float* __restrict__ s_x_ptr,
                                     unsigned int* __restrict__ dst,
                                     int n4, int apply_scale) {
  int i = blockIdx.x * blockDim.x + threadIdx.x;
  if (i >= n4) return;
  const float inv = apply_scale ? (1.0f / s_x_ptr[0]) : 1.0f;
  const vf4 f = __builtin_nontemporal_load((const vf4*)src + i);
  int b0 = (int)rintf(f[0] * inv);
  int b1 = (int)rintf(f[1] * inv);
  int b2 = (int)rintf(f[2] * inv);
  int b3 = (int)rintf(f[3] * inv);
  b0 = b0 < -128 ? -128 : (b0 > 127 ? 127 : b0);
  b1 = b1 < -128 ? -128 : (b1 > 127 ? 127 : b1);
  b2 = b2 < -128 ? -128 : (b2 > 127 ? 127 : b2);
  b3 = b3 < -128 ? -128 : (b3 > 127 ? 127 : b3);
  dst[i] = (unsigned)(b0 & 0xff) | ((unsigned)(b1 & 0xff) << 8) |
           ((unsigned)(b2 & 0xff) << 16) | ((unsigned)(b3 & 0xff) << 24);
}

__global__ __launch_bounds__(NTHREADS) void qgemm_wmma_iu8(
    const signed char* __restrict__ Xq,
    const signed char* __restrict__ Wq,
    const float* __restrict__ bq,
    const float* __restrict__ sa,
    float* __restrict__ out) {
  __shared__ __align__(16) signed char lA[2][BM * BK];
  __shared__ __align__(16) signed char lB[2][BN * BK];

  const int tid = threadIdx.x;
  const int lane = tid & 31;
  const int wave = tid >> 5;
  const int waveM = wave >> 2;
  const int waveN = wave & 3;

  const int mBase = blockIdx.x * BM;
  const int nBase = blockIdx.y * BN;

  v8i acc[4][2];
  for (int m = 0; m < 4; ++m)
    for (int n = 0; n < 2; ++n)
      acc[m][n] = (v8i){0, 0, 0, 0, 0, 0, 0, 0};

  const int l15 = lane & 15;
  const int lHi = lane >> 4;
  const int aKoff = lHi * 16;
  const int bKoff = lHi * 16;

  const int srow = tid >> 2;
  const int scol = (tid & 3) * 16;
  const signed char* gA = Xq + (size_t)(mBase + srow) * K_DIM + scol;
  const signed char* gB = Wq + (size_t)(nBase + srow) * K_DIM + scol;
  const int lOff0 = srow * BK + scol;
  const int lOff1 = (srow + 64) * BK + scol;

  auto stage = [&](int buf, int k0) {
    copy16_g2l(gA + k0, &lA[buf][lOff0]);
    copy16_g2l(gA + k0 + (size_t)64 * K_DIM, &lA[buf][lOff1]);
    copy16_g2l(gB + k0, &lB[buf][lOff0]);
    copy16_g2l(gB + k0 + (size_t)64 * K_DIM, &lB[buf][lOff1]);
  };

  auto compute = [&](int buf) {
    v8i afrag[4];
    for (int m = 0; m < 4; ++m) {
      const signed char* ap = &lA[buf][(waveM * 64 + m * 16 + l15) * BK + aKoff];
      const int2 t0 = *(const int2*)(ap + 0);
      const int2 t1 = *(const int2*)(ap + 8);
      const int2 t2 = *(const int2*)(ap + 32);
      const int2 t3 = *(const int2*)(ap + 40);
      v8i a;
      a[0] = t0.x; a[1] = t0.y; a[2] = t1.x; a[3] = t1.y;
      a[4] = t2.x; a[5] = t2.y; a[6] = t3.x; a[7] = t3.y;
      afrag[m] = a;
    }
    for (int n = 0; n < 2; ++n) {
      const signed char* bp = &lB[buf][(waveN * 32 + n * 16 + l15) * BK + bKoff];
      const int4 u0 = *(const int4*)(bp + 0);
      const int4 u1 = *(const int4*)(bp + 32);
      v8i b;
      b[0] = u0.x; b[1] = u0.y; b[2] = u0.z; b[3] = u0.w;
      b[4] = u1.x; b[5] = u1.y; b[6] = u1.z; b[7] = u1.w;
      for (int m = 0; m < 4; ++m) {
        acc[m][n] = __builtin_amdgcn_wmma_i32_16x16x64_iu8(
             true, afrag[m],  true, b, acc[m][n],
             false,  false);
      }
    }
  };

  stage(0, 0);
  wait_stage();
#pragma unroll
  for (int k = 0; k < KSTEPS - 1; ++k) {
    stage((k + 1) & 1, (k + 1) * BK);
    compute(k & 1);
    wait_stage();
  }
  compute((KSTEPS - 1) & 1);

  for (int n = 0; n < 2; ++n) {
    const int gN = nBase + waveN * 32 + n * 16 + l15;
    const float bb = bq[gN];
    const float ss = sa[gN];
    for (int m = 0; m < 4; ++m) {
      const int gM0 = mBase + waveM * 64 + m * 16 + lHi * 8;
      const v8i a = acc[m][n];
      for (int r = 0; r < 8; ++r) {
        __builtin_nontemporal_store(((float)a[r] + bb) * ss,
                                    out + (size_t)(gM0 + r) * N_DIM + gN);
      }
    }
  }
}

extern "C" void kernel_launch(void* const* d_in, const int* in_sizes, int n_in,
                              void* d_out, int out_size, void* d_ws, size_t ws_size,
                              hipStream_t stream) {
  const float* x_hat  = (const float*)d_in[0];
  const float* s_x    = (const float*)d_in[1];
  const float* w_f32  = (const float*)d_in[2];
  const float* s_w    = (const float*)d_in[3];
  const float* b_int8 = (const float*)d_in[4];
  float* out = (float*)d_out;

  signed char* Xq = (signed char*)d_ws;
  signed char* Wq = Xq + (size_t)M_TOT * K_DIM;
  float* bqd = (float*)(Wq + (size_t)N_DIM * K_DIM);
  float* sa = out + (size_t)M_TOT * N_DIM;

  prep_scales<<<(N_DIM + 255) / 256, 256, 0, stream>>>(s_w, b_int8, s_x, sa, bqd);

  {
    int n4 = (N_DIM * K_DIM) / 4;
    quant_pack_f32_to_i8<<<(n4 + 255) / 256, 256, 0, stream>>>(
        w_f32, s_x, (unsigned int*)Wq, n4,  0);
  }
  {
    int n4 = (M_TOT * K_DIM) / 4;
    quant_pack_f32_to_i8<<<(n4 + 255) / 256, 256, 0, stream>>>(
        x_hat, s_x, (unsigned int*)Xq, n4,  1);
  }

  dim3 grid(M_TOT / BM, N_DIM / BN);
  qgemm_wmma_iu8<<<grid, NTHREADS, 0, stream>>>(Xq, Wq, bqd, sa, out);
}
